// UEC2DTA_77421080477774
// MI455X (gfx1250) — matrix operand order repaired and checked by reading, not compiled, not run on hardware
//
#include <hip/hip_runtime.h>

typedef __bf16 bf16_t;
typedef __bf16 v16bf __attribute__((ext_vector_type(16)));
typedef __bf16 v8bf  __attribute__((ext_vector_type(8)));
typedef float  v8f   __attribute__((ext_vector_type(8)));
typedef float  v4f   __attribute__((ext_vector_type(4)));
typedef int    v4i   __attribute__((ext_vector_type(4)));

#define N_NODES 8192
#define DIM     512

#define AS1 __attribute__((address_space(1)))
#define AS3 __attribute__((address_space(3)))

#if __has_builtin(__builtin_amdgcn_global_load_async_to_lds_b128)
#define USE_ASYNC 1
#else
#define USE_ASYNC 0
#endif

static __device__ __forceinline__ void cp_lds16(const bf16_t* g, bf16_t* l) {
#if USE_ASYNC
    __builtin_amdgcn_global_load_async_to_lds_b128((AS1 v4i*)g, (AS3 v4i*)l, 0, 0);
#else
    *(v8bf*)l = *(const v8bf*)g;   // sync fallback: global load + ds store
#endif
}

static __device__ __forceinline__ void wait_async_done() {
#if USE_ASYNC
#if __has_builtin(__builtin_amdgcn_s_wait_asynccnt)
    __builtin_amdgcn_s_wait_asynccnt(0);
#else
    asm volatile("s_wait_asynccnt 0" ::: "memory");
#endif
#endif
}

// ---------------- conversion / transpose helpers ----------------

__global__ void k_f32_to_bf16(const float* __restrict__ in, bf16_t* __restrict__ out, int n) {
    int i = blockIdx.x * blockDim.x + threadIdx.x;
    if (i < n) out[i] = (bf16_t)in[i];
}

// WT[n][k] = W[k][n] as bf16, 512x512
__global__ void k_transpose_w_bf16(const float* __restrict__ W, bf16_t* __restrict__ WT) {
    int idx = blockIdx.x * blockDim.x + threadIdx.x;
    int n = idx >> 9, k = idx & 511;
    WT[n * DIM + k] = (bf16_t)W[k * DIM + n];
}

// HT[c][r] = H[r][c], H: [8192,512] bf16 -> HT: [512,8192] bf16 (LDS-tiled)
__global__ void k_transpose_h(const bf16_t* __restrict__ H, bf16_t* __restrict__ HT) {
    __shared__ bf16_t tile[32][33];
    int tx = threadIdx.x, ty = threadIdx.y;          // 32 x 8
    int cb = blockIdx.x * 32;
    int rb = blockIdx.y * 32;
#pragma unroll
    for (int i = 0; i < 32; i += 8)
        tile[ty + i][tx] = H[(size_t)(rb + ty + i) * DIM + cb + tx];
    __syncthreads();
#pragma unroll
    for (int i = 0; i < 32; i += 8)
        HT[(size_t)(cb + ty + i) * N_NODES + rb + tx] = tile[tx][ty + i];
}

// ---------------- projection GEMM: C = A @ W (via WT), bf16 in, bf16 out ----------------
// One wave computes a 16x128 tile; per k-step all 8 B tiles are preloaded so the
// loads issue as one clause and overlap the WMMA burst.

__global__ void __launch_bounds__(32) k_gemm_bf16(const bf16_t* __restrict__ A,   // [8192,512]
                                                  const bf16_t* __restrict__ WT,  // [512,512]
                                                  bf16_t* __restrict__ C) {       // [8192,512]
    const int lane = threadIdx.x & 31;
    const int r = lane & 15, half = lane >> 4;
    const int mbase = (blockIdx.x >> 2) * 16;
    const int nbase = (blockIdx.x & 3) * 128;

    v8f c[8];
#pragma unroll
    for (int t = 0; t < 8; ++t) c[t] = v8f{0.f, 0.f, 0.f, 0.f, 0.f, 0.f, 0.f, 0.f};

    const bf16_t* arow = A + (size_t)(mbase + r) * DIM;
    const bf16_t* brow = WT + (size_t)(nbase + r) * DIM + half * 16;
    for (int k = 0; k < DIM; k += 32) {
        v8bf alo = *(const v8bf*)(arow + k + half * 16);
        v8bf ahi = *(const v8bf*)(arow + k + half * 16 + 8);
        v16bf a = __builtin_shufflevector(alo, ahi, 0, 1, 2, 3, 4, 5, 6, 7, 8, 9, 10, 11, 12, 13, 14, 15);
        v16bf b[8];
#pragma unroll
        for (int t = 0; t < 8; ++t)
            b[t] = *(const v16bf*)(brow + (size_t)t * 16 * DIM + k);
#pragma unroll
        for (int t = 0; t < 8; ++t)
            c[t] = __builtin_amdgcn_wmma_f32_16x16x32_bf16(false, a, false, b[t], (short)0, c[t], false, false);
    }
#pragma unroll
    for (int t = 0; t < 8; ++t) {
#pragma unroll
        for (int v = 0; v < 8; ++v) {
            int row = mbase + v + half * 8;
            int col = nbase + t * 16 + r;
            C[(size_t)row * DIM + col] = (bf16_t)c[t][v];
        }
    }
}

// ---------------- GEMVs: hd[i] = h_i . adst, hs[i] = h_i . asrc ----------------

__global__ void k_gemv2(const bf16_t* __restrict__ H, const float* __restrict__ adst,
                        const float* __restrict__ asrc, float* __restrict__ hd,
                        float* __restrict__ hs) {
    int wave = threadIdx.x >> 5;
    int lane = threadIdx.x & 31;
    int row = blockIdx.x * 4 + wave;
    const bf16_t* hrow = H + (size_t)row * DIM;
    float sd = 0.f, ss = 0.f;
#pragma unroll 4
    for (int t = 0; t < DIM / 32; ++t) {
        float h = (float)hrow[lane + t * 32];
        sd += h * adst[lane + t * 32];
        ss += h * asrc[lane + t * 32];
    }
#pragma unroll
    for (int off = 16; off; off >>= 1) {
        sd += __shfl_xor(sd, off, 32);
        ss += __shfl_xor(ss, off, 32);
    }
    if (lane == 0) { hd[row] = sd; hs[row] = ss; }
}

// ---------------- fused masked-softmax attention ----------------
// 4 waves/block, 64 dst rows/block. The shared 32x512 h^T tile for each j-step is
// staged into LDS with async-to-LDS copies (double-buffered, one barrier per step);
// each wave runs an online softmax over its 16 rows and 32 WMMA accum tiles.

#define WAVES       4
#define ROWS_BLK    (16 * WAVES)
#define LSTRIDE     40                       // 32 data + 8 pad bf16 per LDS row
#define JTILES      (N_NODES / 32)

__global__ void __launch_bounds__(32 * WAVES) k_attn(
        const float* __restrict__ hd, const float* __restrict__ hs,
        const float* __restrict__ adj, const bf16_t* __restrict__ HT,  // [512, 8192]
        const float* __restrict__ bias, bf16_t* __restrict__ outb,
        float* __restrict__ outf) {
    const int tid  = threadIdx.x;
    const int wave = tid >> 5, lane = tid & 31;
    const int r = lane & 15, half = lane >> 4;
    const int Kb = half * 8;
    const int mbase = blockIdx.x * ROWS_BLK + wave * 16;
    const int rowg = mbase + r;

    __shared__ bf16_t lbuf[2][DIM * LSTRIDE];           // 2 x 40KB

    v8f acc[32];
#pragma unroll
    for (int t = 0; t < 32; ++t) acc[t] = v8f{0.f, 0.f, 0.f, 0.f, 0.f, 0.f, 0.f, 0.f};

    float m = -1e30f, l = 0.f;
    const float hdr = hd[rowg];
    const float* adjrow = adj + (size_t)rowg * N_NODES;

    // stage the 512-row x 64-byte tile for column block jb into lbuf[b]
    auto stage = [&](int jb, int b) {
#pragma unroll
        for (int q = 0; q < 2048 / (32 * WAVES); ++q) {   // 2048 16-byte units
            int unit = q * (32 * WAVES) + tid;
            int row = unit >> 2, seg = unit & 3;
            cp_lds16(HT + (size_t)row * N_NODES + jb + seg * 8,
                     &lbuf[b][row * LSTRIDE + seg * 8]);
        }
    };

    stage(0, 0);
    wait_async_done();
    __syncthreads();

    for (int jt = 0; jt < JTILES; ++jt) {
        const int jb = jt * 32;
        const int cur = jt & 1;
        if (jt + 1 < JTILES) stage(jb + 32, cur ^ 1);

        float sv[16], av[16];
        ((v4f*)sv)[0] = *(const v4f*)(hs + jb + Kb);
        ((v4f*)sv)[1] = *(const v4f*)(hs + jb + Kb + 4);
        ((v4f*)sv)[2] = *(const v4f*)(hs + jb + Kb + 16);
        ((v4f*)sv)[3] = *(const v4f*)(hs + jb + Kb + 20);
        ((v4f*)av)[0] = *(const v4f*)(adjrow + jb + Kb);
        ((v4f*)av)[1] = *(const v4f*)(adjrow + jb + Kb + 4);
        ((v4f*)av)[2] = *(const v4f*)(adjrow + jb + Kb + 16);
        ((v4f*)av)[3] = *(const v4f*)(adjrow + jb + Kb + 20);

        float e[16], tm = -1e30f;
#pragma unroll
        for (int i = 0; i < 16; ++i) {
            int jcol = jb + Kb + (i < 8 ? i : i + 8);
            float ev = hdr + sv[i];
            ev = ev > 0.f ? ev : 0.2f * ev;                   // leaky_relu(0.2)
            bool edge = (av[i] != 0.f) || (jcol == rowg);     // adj mask + self loop
            e[i] = edge ? ev : -1e30f;
            tm = fmaxf(tm, e[i]);
        }
        tm = fmaxf(tm, __shfl_xor(tm, 16, 32));
        float mnew = fmaxf(m, tm);

        float p[16], ps = 0.f;
#pragma unroll
        for (int i = 0; i < 16; ++i) { p[i] = __expf(e[i] - mnew); ps += p[i]; }
        ps += __shfl_xor(ps, 16, 32);
        float scale = __expf(m - mnew);

        if (__ballot(mnew > m) != 0ull) {
            float rs[8];
#pragma unroll
            for (int v = 0; v < 8; ++v) rs[v] = __shfl(scale, v + Kb, 32);
#pragma unroll
            for (int t = 0; t < 32; ++t)
#pragma unroll
                for (int v = 0; v < 8; ++v) acc[t][v] *= rs[v];
        }
        l = l * scale + ps;
        m = mnew;

        v16bf pa;
#pragma unroll
        for (int i = 0; i < 16; ++i) pa[i] = (bf16_t)p[i];

        const bf16_t* base = &lbuf[cur][0];
#pragma unroll
        for (int t = 0; t < 32; ++t) {
            const bf16_t* bp = base + (t * 16 + r) * LSTRIDE + half * 8;
            v8bf blo = *(const v8bf*)bp;
            v8bf bhi = *(const v8bf*)(bp + 16);
            v16bf b = __builtin_shufflevector(blo, bhi, 0, 1, 2, 3, 4, 5, 6, 7, 8, 9, 10, 11, 12, 13, 14, 15);
            acc[t] = __builtin_amdgcn_wmma_f32_16x16x32_bf16(false, pa, false, b, (short)0, acc[t], false, false);
        }

        wait_async_done();        // next tile's async copies landed
        __syncthreads();          // and everyone is done reading lbuf[cur]
    }

    float linv[8];
#pragma unroll
    for (int v = 0; v < 8; ++v) linv[v] = 1.f / __shfl(l, v + Kb, 32);

#pragma unroll
    for (int t = 0; t < 32; ++t) {
        int d = t * 16 + r;
        float bval = bias[d];
#pragma unroll
        for (int v = 0; v < 8; ++v) {
            int row = mbase + v + half * 8;
            float val = fmaxf(acc[t][v] * linv[v] + bval, 0.f);
            if (outf) outf[(size_t)row * DIM + d] = val;
            else      outb[(size_t)row * DIM + d] = (bf16_t)val;
        }
    }
}

// ---------------- driver ----------------

extern "C" void kernel_launch(void* const* d_in, const int* in_sizes, int n_in,
                              void* d_out, int out_size, void* d_ws, size_t ws_size,
                              hipStream_t stream) {
    const float* x     = (const float*)d_in[0];
    const float* adj   = (const float*)d_in[1];
    const float* W1    = (const float*)d_in[2];
    const float* b1    = (const float*)d_in[3];
    const float* asrc1 = (const float*)d_in[4];
    const float* adst1 = (const float*)d_in[5];
    const float* W2    = (const float*)d_in[6];
    const float* b2    = (const float*)d_in[7];
    const float* asrc2 = (const float*)d_in[8];
    const float* adst2 = (const float*)d_in[9];

    char* ws = (char*)d_ws;
    size_t off = 0;
    bf16_t* xb  = (bf16_t*)(ws + off); off += (size_t)N_NODES * DIM * 2;  // reused as o1b
    bf16_t* hb  = (bf16_t*)(ws + off); off += (size_t)N_NODES * DIM * 2;
    bf16_t* hT  = (bf16_t*)(ws + off); off += (size_t)N_NODES * DIM * 2;
    bf16_t* W1T = (bf16_t*)(ws + off); off += (size_t)DIM * DIM * 2;
    bf16_t* W2T = (bf16_t*)(ws + off); off += (size_t)DIM * DIM * 2;
    float*  hd  = (float*)(ws + off);  off += (size_t)N_NODES * 4;
    float*  hs  = (float*)(ws + off);  off += (size_t)N_NODES * 4;

    dim3 tb(32, 8), tg(DIM / 32, N_NODES / 32);

    // ---- layer 1 ----
    k_f32_to_bf16<<<(N_NODES * DIM) / 256, 256, 0, stream>>>(x, xb, N_NODES * DIM);
    k_transpose_w_bf16<<<(DIM * DIM) / 256, 256, 0, stream>>>(W1, W1T);
    k_transpose_w_bf16<<<(DIM * DIM) / 256, 256, 0, stream>>>(W2, W2T);
    k_gemm_bf16<<<(N_NODES / 16) * (DIM / 128), 32, 0, stream>>>(xb, W1T, hb);
    k_gemv2<<<N_NODES / 4, 128, 0, stream>>>(hb, adst1, asrc1, hd, hs);
    k_transpose_h<<<tg, tb, 0, stream>>>(hb, hT);
    k_attn<<<N_NODES / ROWS_BLK, 32 * WAVES, 0, stream>>>(hd, hs, adj, hT, b1, xb, nullptr);

    // ---- layer 2 ----
    k_gemm_bf16<<<(N_NODES / 16) * (DIM / 128), 32, 0, stream>>>(xb, W2T, hb);
    k_gemv2<<<N_NODES / 4, 128, 0, stream>>>(hb, adst2, asrc2, hd, hs);
    k_transpose_h<<<tg, tb, 0, stream>>>(hb, hT);
    k_attn<<<N_NODES / ROWS_BLK, 32 * WAVES, 0, stream>>>(hd, hs, adj, hT, b2, nullptr, (float*)d_out);
}
